// MSTDependencyParser_46643344834695
// MI455X (gfx1250) — compile-verified, matrix operand order checked by reading, not run on hardware
//
#include <hip/hip_runtime.h>

// ---------------------------------------------------------------------------
// MST dependency parser forward pass for gfx1250 (MI455X).
//   seq = concat(words, pos_emb[poss])            [1024 x 325]
//   v1  = BiLSTM0(seq)    v2 = BiLSTM1(v1)        [1024 x 250]
//   hh  = v2 @ fc1_w[:, :250].T + fc1_b ; hm = v2 @ fc1_w[:, 250:].T
//   scores[i,j] = fc2_b + sum_f fc2_w[f] * tanh(hm[i,f] + hh[j,f])
//   loss = CE(scores, parent_ids)
// GEMMs -> v_wmma_f32_16x16x32_f16 ; nonlinearities -> v_tanh_f32 (TRANS).
// ---------------------------------------------------------------------------

typedef __attribute__((ext_vector_type(16))) _Float16 v16h;
typedef __attribute__((ext_vector_type(8)))  _Float16 v8h;
typedef __attribute__((ext_vector_type(8)))  float    v8f;

#define NSEQ 1024
#define HN   125
#define FCN  100

union Frag { v16h v; v8h h[2]; };

// CDNA5 transcendental tanh; v_nop covers the TRANS co-execution hazard
// (CDNA5 ISA: 1 independent op / V_NOP after a trans op).
__device__ __forceinline__ float ftanh(float x) {
  float y;
  asm volatile("v_tanh_f32 %0, %1\n\tv_nop" : "=v"(y) : "v"(x));
  return y;
}
__device__ __forceinline__ float fsig(float x) {
  return 0.5f * ftanh(0.5f * x) + 0.5f;
}

// ---------------------------------------------------------------------------
// Packing / staging kernels (f32 -> padded f16 operand buffers)
// ---------------------------------------------------------------------------
__global__ void zero_h16(_Float16* p, int n) {
  int i = blockIdx.x * blockDim.x + threadIdx.x;
  if (i < n) p[i] = (_Float16)0.f;
}

__global__ void pack_seq(const float* __restrict__ words,
                         const float* __restrict__ pos_emb,
                         const int*   __restrict__ poss,
                         _Float16* __restrict__ out) {       // [1024 x 352]
  int t = blockIdx.x;
  int c = threadIdx.x;                                       // 0..351
  float v = 0.f;
  if (c < 300)      v = words[t * 300 + c];
  else if (c < 325) v = pos_emb[poss[t] * 25 + (c - 300)];
  out[t * 352 + c] = (_Float16)v;
}

// LSTM input-proj weights [2, 500, K] -> [1024 rows x Kpad] f16 (dir d at
// row d*512), plus fused bias (b_ih + b_hh) -> [1024] f32.
__global__ void pack_wlstm(const float* __restrict__ wih,
                           const float* __restrict__ bih,
                           const float* __restrict__ bhh,
                           _Float16* __restrict__ wout,
                           float* __restrict__ bout, int K, int Kpad) {
  int rowp = blockIdx.x;            // 0..1023
  int c    = threadIdx.x;           // 0..Kpad-1
  int dir  = rowp >> 9;
  int g    = rowp & 511;
  bool valid = g < 500;
  float v = (valid && c < K) ? wih[(size_t)(dir * 500 + g) * K + c] : 0.f;
  wout[(size_t)rowp * Kpad + c] = (_Float16)v;
  if (c == 0)
    bout[rowp] = valid ? (bih[dir * 500 + g] + bhh[dir * 500 + g]) : 0.f;
}

// fc1_w [100 x 500] -> stacked [256 x 256] f16: rows 0..99 = head half
// (cols 0..249), rows 128..227 = modifier half (cols 250..499).
// fc1_b folded only into head rows (pre = hm + hh + b -> add b once to hh).
__global__ void pack_wf(const float* __restrict__ fc1w,
                        const float* __restrict__ fc1b,
                        _Float16* __restrict__ wout,
                        float* __restrict__ bout) {
  int r = blockIdx.x;               // 0..255
  int c = threadIdx.x;              // 0..255
  float v = 0.f;
  if (r < FCN && c < 250)                     v = fc1w[r * 500 + c];
  else if (r >= 128 && r < 128 + FCN && c < 250) v = fc1w[(r - 128) * 500 + 250 + c];
  wout[r * 256 + c] = (_Float16)v;
  if (c == 0) bout[r] = (r < FCN) ? fc1b[r] : 0.f;
}

// ---------------------------------------------------------------------------
// WMMA GEMM:  C[M x Nout] = A[M x K] * B[Nout x K]^T + bias, f16 in / f32 out.
// Block = 128 threads (4 waves); each wave owns a 16(M) x 64(N) tile.
// Fragment loads follow the ISA 16-bit A/B layout: lane l -> row (l&15),
// halfs 0..7 = K[k + (l>>4)*8 ..], halfs 8..15 = K[k+16 + (l>>4)*8 ..].
// ---------------------------------------------------------------------------
__global__ __launch_bounds__(128) void wmma_gemm_nt(
    const _Float16* __restrict__ A, const _Float16* __restrict__ B,
    const float* __restrict__ bias, float* __restrict__ C, int K, int ldc) {
  const int lane = threadIdx.x & 31;
  const int wave = threadIdx.x >> 5;
  const int m0 = blockIdx.y * 64 + wave * 16;
  const int n0 = blockIdx.x * 64;
  const int r  = lane & 15;
  const int hi = lane >> 4;

  v8f acc0 = {}; v8f acc1 = {}; v8f acc2 = {}; v8f acc3 = {};
  const _Float16* arow  = A + (size_t)(m0 + r) * K;
  const _Float16* brow0 = B + (size_t)(n0 + r) * K;
  const _Float16* brow1 = brow0 + (size_t)16 * K;
  const _Float16* brow2 = brow0 + (size_t)32 * K;
  const _Float16* brow3 = brow0 + (size_t)48 * K;

  for (int k = 0; k < K; k += 32) {
    const int o0 = k + hi * 8;
    const int o1 = k + 16 + hi * 8;
    Frag a, b0, b1, b2, b3;
    a.h[0]  = *(const v8h*)(arow  + o0);  a.h[1]  = *(const v8h*)(arow  + o1);
    b0.h[0] = *(const v8h*)(brow0 + o0);  b0.h[1] = *(const v8h*)(brow0 + o1);
    b1.h[0] = *(const v8h*)(brow1 + o0);  b1.h[1] = *(const v8h*)(brow1 + o1);
    b2.h[0] = *(const v8h*)(brow2 + o0);  b2.h[1] = *(const v8h*)(brow2 + o1);
    b3.h[0] = *(const v8h*)(brow3 + o0);  b3.h[1] = *(const v8h*)(brow3 + o1);
    acc0 = __builtin_amdgcn_wmma_f32_16x16x32_f16(false, a.v, false, b0.v, (short)0, acc0, false, false);
    acc1 = __builtin_amdgcn_wmma_f32_16x16x32_f16(false, a.v, false, b1.v, (short)0, acc1, false, false);
    acc2 = __builtin_amdgcn_wmma_f32_16x16x32_f16(false, a.v, false, b2.v, (short)0, acc2, false, false);
    acc3 = __builtin_amdgcn_wmma_f32_16x16x32_f16(false, a.v, false, b3.v, (short)0, acc3, false, false);
  }

  float bb0 = bias ? bias[n0 + r]      : 0.f;
  float bb1 = bias ? bias[n0 + 16 + r] : 0.f;
  float bb2 = bias ? bias[n0 + 32 + r] : 0.f;
  float bb3 = bias ? bias[n0 + 48 + r] : 0.f;
#pragma unroll
  for (int v = 0; v < 8; ++v) {
    int row = m0 + hi * 8 + v;          // C/D layout: VGPR v -> M = v + hi*8
    size_t off = (size_t)row * ldc + n0 + r;
    C[off +  0] = acc0[v] + bb0;
    C[off + 16] = acc1[v] + bb1;
    C[off + 32] = acc2[v] + bb2;
    C[off + 48] = acc3[v] + bb3;
  }
}

// ---------------------------------------------------------------------------
// LSTM scan: one workgroup per direction (grid.x = 2). xi already holds
// x@Wih^T + bih + bhh at col offset dir*512. h is written as f16 into
// vout[t, dir*125 + j] (next-layer GEMM operand).
// ---------------------------------------------------------------------------
__global__ __launch_bounds__(512) void lstm_scan(
    const float* __restrict__ xi,      // [1024 x 1024]
    const float* __restrict__ whh,     // [2 x 500 x 125]
    _Float16* __restrict__ vout) {     // [1024 x 256]
  const int dir = blockIdx.x;
  const int j   = threadIdx.x;
  __shared__ float sh[128];
  __shared__ float sg[512];
  if (j < HN) sh[j] = 0.f;
  float c = 0.f;
  __syncthreads();
  const float* W = whh + (size_t)dir * 500 * HN;
  for (int t = 0; t < NSEQ; ++t) {
    const int tt = dir ? (NSEQ - 1 - t) : t;
    if (j < 500) {
      float acc = xi[(size_t)tt * 1024 + dir * 512 + j];
      const float* wr = W + j * HN;
      for (int k = 0; k < HN; ++k) acc += wr[k] * sh[k];
      sg[j] = acc;
    }
    __syncthreads();
    if (j < HN) {
      float ig = fsig(sg[j]);
      float fg = fsig(sg[HN + j]);
      float gg = ftanh(sg[2 * HN + j]);
      float og = fsig(sg[3 * HN + j]);
      c = fg * c + ig * gg;
      float hn = og * ftanh(c);
      sh[j] = hn;
      vout[(size_t)tt * 256 + dir * HN + j] = (_Float16)hn;
    }
    __syncthreads();
  }
}

// ---------------------------------------------------------------------------
// scores[i,j] = fc2_b + sum_f fc2_w[f] * tanh(hm[i,f] + hh[j,f])
// (fc1_b already folded into hh). 32x32 tile per 256-thread block.
// ---------------------------------------------------------------------------
__global__ __launch_bounds__(256) void score_kernel(
    const float* __restrict__ hhm,     // [1024 x 256]: hh @ col 0, hm @ col 128
    const float* __restrict__ fc2_w, const float* __restrict__ fc2_b,
    float* __restrict__ scores) {      // [1024 x 1024]
  __shared__ float s_hh[32][FCN];
  __shared__ float s_hm[32][FCN];
  __shared__ float s_w[FCN];
  const int i0 = blockIdx.y * 32;
  const int j0 = blockIdx.x * 32;
  const int tid = threadIdx.x;
  for (int x = tid; x < 32 * FCN; x += 256) {
    int rr = x / FCN, ff = x - rr * FCN;
    s_hh[rr][ff] = hhm[(size_t)(j0 + rr) * 256 + ff];
    s_hm[rr][ff] = hhm[(size_t)(i0 + rr) * 256 + 128 + ff];
  }
  if (tid < FCN) s_w[tid] = fc2_w[tid];
  __syncthreads();
  const float b2 = fc2_b[0];
  const int il = tid >> 3;             // local i (0..31)
  const int jb = (tid & 7) * 4;        // 4 consecutive local j per thread
  float a0 = 0.f, a1 = 0.f, a2 = 0.f, a3 = 0.f;
  for (int f = 0; f < FCN; ++f) {
    float m = s_hm[il][f];
    float w = s_w[f];
    a0 += w * ftanh(m + s_hh[jb + 0][f]);
    a1 += w * ftanh(m + s_hh[jb + 1][f]);
    a2 += w * ftanh(m + s_hh[jb + 2][f]);
    a3 += w * ftanh(m + s_hh[jb + 3][f]);
  }
  size_t base = (size_t)(i0 + il) * NSEQ + j0 + jb;
  scores[base + 0] = a0 + b2;
  scores[base + 1] = a1 + b2;
  scores[base + 2] = a2 + b2;
  scores[base + 3] = a3 + b2;
}

// ---------------------------------------------------------------------------
// Cross-entropy: one block per row; -mean_i (scores[i,p_i] - logsumexp_j)
// ---------------------------------------------------------------------------
__global__ void loss_init(float* acc) { if (threadIdx.x == 0) *acc = 0.f; }

__global__ __launch_bounds__(256) void row_ce(const float* __restrict__ scores,
                                              const int* __restrict__ parent,
                                              float* __restrict__ acc) {
  const int i = blockIdx.x;
  const int tid = threadIdx.x;
  __shared__ float red[256];
  const float* row = scores + (size_t)i * NSEQ;
  float mx = -3.4e38f;
  for (int j = tid; j < NSEQ; j += 256) mx = fmaxf(mx, row[j]);
  red[tid] = mx; __syncthreads();
  for (int s = 128; s > 0; s >>= 1) {
    if (tid < s) red[tid] = fmaxf(red[tid], red[tid + s]);
    __syncthreads();
  }
  const float m = red[0];
  __syncthreads();
  float sum = 0.f;
  for (int j = tid; j < NSEQ; j += 256) sum += expf(row[j] - m);
  red[tid] = sum; __syncthreads();
  for (int s = 128; s > 0; s >>= 1) {
    if (tid < s) red[tid] += red[tid + s];
    __syncthreads();
  }
  if (tid == 0) {
    float lse = m + logf(red[0]);
    atomicAdd(acc, lse - row[parent[i]]);
  }
}

__global__ void loss_final(const float* acc, float* out) {
  if (threadIdx.x == 0) out[0] = *acc / (float)NSEQ;
}

// ---------------------------------------------------------------------------
// Workspace layout (bytes, buffers reused across sequential stages)
// ---------------------------------------------------------------------------
static constexpr size_t OFF_SEQ  = 0;                         // f16 1024x352
static constexpr size_t OFF_W    = OFF_SEQ  + 1024 * 352 * 2; // f16 w0/w1/wf
static constexpr size_t OFF_XI   = OFF_W    + 1024 * 352 * 2; // f32 1024x1024
static constexpr size_t OFF_BIAS = OFF_XI   + 1024 * 1024 * 4;// f32 1024
static constexpr size_t OFF_VH   = OFF_BIAS + 1024 * 4;       // f16 1024x256
static constexpr size_t OFF_HHM  = OFF_VH   + 1024 * 256 * 2; // f32 1024x256
static constexpr size_t OFF_ACC  = OFF_HHM  + 1024 * 256 * 4; // f32 1

extern "C" void kernel_launch(void* const* d_in, const int* in_sizes, int n_in,
                              void* d_out, int out_size, void* d_ws, size_t ws_size,
                              hipStream_t stream) {
  (void)in_sizes; (void)n_in; (void)out_size; (void)ws_size;
  const float* words   = (const float*)d_in[0];
  const float* pos_emb = (const float*)d_in[1];
  const float* w_ih_l0 = (const float*)d_in[2];
  const float* w_hh_l0 = (const float*)d_in[3];
  const float* b_ih_l0 = (const float*)d_in[4];
  const float* b_hh_l0 = (const float*)d_in[5];
  const float* w_ih_l1 = (const float*)d_in[6];
  const float* w_hh_l1 = (const float*)d_in[7];
  const float* b_ih_l1 = (const float*)d_in[8];
  const float* b_hh_l1 = (const float*)d_in[9];
  const float* fc1_w   = (const float*)d_in[10];
  const float* fc1_b   = (const float*)d_in[11];
  const float* fc2_w   = (const float*)d_in[12];
  const float* fc2_b   = (const float*)d_in[13];
  const int*   poss    = (const int*)d_in[14];
  const int*   parent  = (const int*)d_in[15];

  char* ws = (char*)d_ws;
  _Float16* seq_h = (_Float16*)(ws + OFF_SEQ);
  _Float16* w_h   = (_Float16*)(ws + OFF_W);
  float*    xi    = (float*)   (ws + OFF_XI);
  float*    biasb = (float*)   (ws + OFF_BIAS);
  _Float16* v_h   = (_Float16*)(ws + OFF_VH);
  float*    hhm   = (float*)   (ws + OFF_HHM);
  float*    acc   = (float*)   (ws + OFF_ACC);

  float* lossp  = (float*)d_out;
  float* scores = (float*)d_out + 1;

  // Zero v_h once (scan only writes cols 0..249; pads must stay 0 for WMMA K).
  zero_h16<<<dim3((1024 * 256 + 255) / 256), dim3(256), 0, stream>>>(v_h, 1024 * 256);
  loss_init<<<dim3(1), dim3(32), 0, stream>>>(acc);

  // Stage seq = concat(words, pos_emb[poss]) as f16 [1024 x 352]
  pack_seq<<<dim3(NSEQ), dim3(352), 0, stream>>>(words, pos_emb, poss, seq_h);

  // ---- Layer 0 ----
  pack_wlstm<<<dim3(1024), dim3(352), 0, stream>>>(w_ih_l0, b_ih_l0, b_hh_l0,
                                                   w_h, biasb, 325, 352);
  wmma_gemm_nt<<<dim3(16, 16), dim3(128), 0, stream>>>(seq_h, w_h, biasb, xi, 352, 1024);
  lstm_scan<<<dim3(2), dim3(512), 0, stream>>>(xi, w_hh_l0, v_h);

  // ---- Layer 1 ----
  pack_wlstm<<<dim3(1024), dim3(256), 0, stream>>>(w_ih_l1, b_ih_l1, b_hh_l1,
                                                   w_h, biasb, 250, 256);
  wmma_gemm_nt<<<dim3(16, 16), dim3(128), 0, stream>>>(v_h, w_h, biasb, xi, 256, 1024);
  lstm_scan<<<dim3(2), dim3(512), 0, stream>>>(xi, w_hh_l1, v_h);

  // ---- fc1 halves (hh | hm) in one GEMM ----
  pack_wf<<<dim3(256), dim3(256), 0, stream>>>(fc1_w, fc1_b, w_h, biasb);
  wmma_gemm_nt<<<dim3(4, 16), dim3(128), 0, stream>>>(v_h, w_h, biasb, hhm, 256, 256);

  // ---- scores + cross-entropy ----
  score_kernel<<<dim3(32, 32), dim3(256), 0, stream>>>(hhm, fc2_w, fc2_b, scores);
  row_ce<<<dim3(NSEQ), dim3(256), 0, stream>>>(scores, parent, acc);
  loss_final<<<dim3(1), dim3(32), 0, stream>>>(acc, lossp);
}
